// KMeans_75282186764733
// MI455X (gfx1250) — matrix operand order repaired and checked by reading, not compiled, not run on hardware
//
#include <hip/hip_runtime.h>
#include <hip/hip_bf16.h>

// ---------------------------------------------------------------------------
// Soft k-means (semi-supervised prototype refinement) for MI455X / gfx1250.
// bf16 WMMA for both GEMMs, fused-DPP (single-instruction) softmax reductions,
// LDS-staged transpose for w^T @ q, f32 atomics for the global reduction.
// ---------------------------------------------------------------------------

typedef __attribute__((ext_vector_type(16))) __bf16 v16bf;
typedef __attribute__((ext_vector_type(8)))  __bf16 v8bf;
typedef __attribute__((ext_vector_type(2)))  __bf16 v2bf;
typedef __attribute__((ext_vector_type(8)))  float  v8f;
typedef __attribute__((ext_vector_type(4)))  float  v4f;
typedef __attribute__((ext_vector_type(4)))  int    v4i;
typedef __attribute__((ext_vector_type(8)))  int    v8i;

#define WMMA_BF16(A, B, Cacc) \
  __builtin_amdgcn_wmma_f32_16x16x32_bf16(false, (A), false, (B), (short)0, (Cacc), false, false)

static constexpr int   KC     = 16;   // n_classes
static constexpr int   KD     = 64;   // feature dim
static constexpr float KBETA  = 5.0f;
static constexpr int   KNITER = 50;
// centroids stored pre-scaled by 2*beta*log2e; ||c||^2 pre-scaled to match,
// so logit (log2 domain) = dot - csq and weights = exp2(logit - max).
static constexpr float KLOG2E = 1.4426950408889634f;
static constexpr float KS     = 2.0f * KBETA * KLOG2E;
static constexpr float KCSQ   = 0.6931471805599453f / (4.0f * KBETA);

// workspace layout (bytes)
static constexpr size_t OFF_ACC_NUM = 0;       // f32[16*64]
static constexpr size_t OFF_ACC_DEN = 4096;    // f32[16]
static constexpr size_t OFF_SUP_SUM = 4160;    // f32[16*64]
static constexpr size_t OFF_COUNTS  = 8256;    // f32[16]
static constexpr size_t OFF_CSQ     = 8320;    // f32[16]
static constexpr size_t OFF_COUNTER = 8384;    // f32[1]
static constexpr size_t OFF_CB      = 8448;    // bf16[16*64]
static constexpr size_t OFF_QB      = 65536;   // bf16[Nq*64]

// ---- DPP16 cross-lane reductions -------------------------------------------
// Butterfly over a 16-lane row: xor1, xor2, half_mirror(i<->7-i),
// mirror(i<->15-i).  Valid for any commutative/associative combine.
// Hot path: DPP fused straight into the binop (1 VALU instruction per step);
// separate in/out operands so the compiler needs no extra copy when the
// input stays live.

__device__ __forceinline__ float rmax16(float x) {
  float r;
  asm("v_max_num_f32_dpp %0, %1, %1 quad_perm:[1,0,3,2] row_mask:0xf bank_mask:0xf bound_ctrl:1\n\t"
      "v_max_num_f32_dpp %0, %0, %0 quad_perm:[2,3,0,1] row_mask:0xf bank_mask:0xf bound_ctrl:1\n\t"
      "v_max_num_f32_dpp %0, %0, %0 row_half_mirror row_mask:0xf bank_mask:0xf bound_ctrl:1\n\t"
      "v_max_num_f32_dpp %0, %0, %0 row_mirror row_mask:0xf bank_mask:0xf bound_ctrl:1"
      : "=v"(r) : "v"(x));
  return r;
}
__device__ __forceinline__ float rsum16(float x) {
  float r;
  asm("v_add_f32_dpp %0, %1, %1 quad_perm:[1,0,3,2] row_mask:0xf bank_mask:0xf bound_ctrl:1\n\t"
      "v_add_f32_dpp %0, %0, %0 quad_perm:[2,3,0,1] row_mask:0xf bank_mask:0xf bound_ctrl:1\n\t"
      "v_add_f32_dpp %0, %0, %0 row_half_mirror row_mask:0xf bank_mask:0xf bound_ctrl:1\n\t"
      "v_add_f32_dpp %0, %0, %0 row_mirror row_mask:0xf bank_mask:0xf bound_ctrl:1"
      : "=v"(r) : "v"(x));
  return r;
}

// Builtin-based DPP helpers for the cold paths (predict / epilogue).
static constexpr int DPP_QUAD_XOR1 = 0xB1;  // quad_perm(1,0,3,2)
static constexpr int DPP_QUAD_XOR2 = 0x4E;  // quad_perm(2,3,0,1)
static constexpr int DPP_HALF_MIRR = 0x141; // row_half_mirror
static constexpr int DPP_ROW_MIRR  = 0x140; // row_mirror

template <int CTRL>
__device__ __forceinline__ float dppf(float x) {
  int r = __builtin_amdgcn_update_dpp(0, __builtin_bit_cast(int, x), CTRL, 0xf, 0xf, true);
  return __builtin_bit_cast(float, r);
}
template <int CTRL>
__device__ __forceinline__ int dppi(int x) {
  return __builtin_amdgcn_update_dpp(0, x, CTRL, 0xf, 0xf, true);
}
template <int CTRL>
__device__ __forceinline__ void argmin_step(float& s, int& iv) {
  float so = dppf<CTRL>(s);
  int   io = dppi<CTRL>(iv);
  if (so < s || (so == s && io < iv)) { s = so; iv = io; }
}
__device__ __forceinline__ int argmin16(float s, int iv) {
  argmin_step<DPP_QUAD_XOR1>(s, iv);
  argmin_step<DPP_QUAD_XOR2>(s, iv);
  argmin_step<DPP_HALF_MIRR>(s, iv);
  argmin_step<DPP_ROW_MIRR>(s, iv);
  return iv;
}

// Build a v16bf A/B fragment from two 16-byte chunks.
__device__ __forceinline__ v16bf mk16(v4i lo, v4i hi) {
  v8i p;
  p[0] = lo[0]; p[1] = lo[1]; p[2] = lo[2]; p[3] = lo[3];
  p[4] = hi[0]; p[5] = hi[1]; p[6] = hi[2]; p[7] = hi[3];
  return __builtin_bit_cast(v16bf, p);
}

// Convert 16 f32 (four float4 chunks) into one v16bf fragment.
__device__ __forceinline__ v16bf cvt16(v4f a, v4f b, v4f c, v4f d) {
  v16bf r;
#pragma unroll
  for (int i = 0; i < 4; ++i) {
    r[i]      = (__bf16)a[i];
    r[4 + i]  = (__bf16)b[i];
    r[8 + i]  = (__bf16)c[i];
    r[12 + i] = (__bf16)d[i];
  }
  return r;
}

// --------------------------- one-time kernels ------------------------------

__global__ void zero_kernel(float* acc_num, float* acc_den, float* counter) {
  int t = threadIdx.x;
  if (t < KC * KD) acc_num[t] = 0.f;
  if (t < KC)      acc_den[t] = 0.f;
  if (t == 0)      *counter   = 0.f;
}

// sup_sum[c][d] and counts[c] from the support rows (deterministic scan).
__global__ __launch_bounds__(1024) void supstats_kernel(
    const float* __restrict__ sf, const int* __restrict__ sl,
    float* __restrict__ sup_sum, float* __restrict__ counts, int ns) {
  int t = threadIdx.x;
  int c = t >> 6, d = t & 63;
  float s = 0.f;
  int cnt = 0;
  for (int i = 0; i < ns; ++i) {
    if (sl[i] == c) { s += sf[(size_t)i * KD + d]; ++cnt; }
  }
  sup_sum[t] = s;
  if (d == 0) counts[c] = (float)cnt;
}

// f32 queries -> bf16 queries (one-time, halves per-iteration HBM traffic).
__global__ void convert_kernel(const float* __restrict__ qf,
                               __bf16* __restrict__ qb, long n8) {
  long i  = (long)blockIdx.x * blockDim.x + threadIdx.x;
  long st = (long)gridDim.x * blockDim.x;
  for (; i < n8; i += st) {
    const v4f* p = (const v4f*)(qf + i * 8);
    v4f f0 = p[0], f1 = p[1];
    v8bf h;
#pragma unroll
    for (int k = 0; k < 4; ++k) {
      h[k]     = (__bf16)f0[k];
      h[4 + k] = (__bf16)f1[k];
    }
    *(v8bf*)(qb + i * 8) = h;
  }
}

// centroid = (sup_sum + acc_num) / (counts + acc_den); emit bf16 scaled by
// 2*beta*log2e plus matching scaled ||c||^2, then zero the accumulators.
__global__ __launch_bounds__(1024) void update_kernel(
    float* __restrict__ acc_num, float* __restrict__ acc_den,
    const float* __restrict__ sup_sum, const float* __restrict__ counts,
    __bf16* __restrict__ cb, float* __restrict__ csq) {
  __shared__ float sq[KC * KD];
  int t = threadIdx.x;
  int c = t >> 6, d = t & 63;
  float num  = sup_sum[t] + acc_num[t];
  float den  = counts[c] + acc_den[c];
  float cent = num / den;
  __bf16 cb16 = (__bf16)(KS * cent);
  cb[t] = cb16;
  float cf = (float)cb16;  // rounded value, consistent with the dot path
  sq[t] = cf * cf;
  acc_num[t] = 0.f;
  __syncthreads();
  if (d == 0) {
    float s = 0.f;
#pragma unroll 8
    for (int i = 0; i < KD; ++i) s += sq[c * KD + i];
    csq[c]     = s * KCSQ;
    acc_den[c] = 0.f;
  }
}

// --------------------------- main iteration kernel -------------------------
//
// Per wave, per step: a pair of 16-query tiles (32 queries).
//   GEMM1 (q @ C^T): v_wmma_f32_16x16x32_bf16, A straight from global
//   softmax over classes via fused v_max/v_add DPP reductions (1 instr/step)
//   GEMM2 (w^T @ q): q-tile transposed through per-wave LDS, w^T packed to
//   LDS, one s_wait_dscnt, then 4x WMMA into persistent v8f accumulators.
template <bool PRE>
__global__ __launch_bounds__(256) void iterate_kernel(
    const float* __restrict__ qf, const __bf16* __restrict__ qb,
    const __bf16* __restrict__ cb, const float* __restrict__ csq_g,
    float* __restrict__ g_num, float* __restrict__ g_den, int nq) {
  __shared__ __bf16 sh_qt[8][KD * 32];  // per-wave q^T  [dim][k]
  __shared__ __bf16 sh_wt[8][KC * 32];  // per-wave w^T  [class][k]
  __shared__ float  s_num[KC * KD];
  __shared__ float  s_den[KC];

  const int tid  = threadIdx.x;
  const int lane = tid & 31;
  const int wv   = tid >> 5;
  const int half = lane >> 4;
  const int l15  = lane & 15;

  for (int i = tid; i < KC * KD; i += blockDim.x) s_num[i] = 0.f;
  if (tid < KC) s_den[tid] = 0.f;
  __syncthreads();

  // Persistent centroid B-fragments: lane n holds column n (K packed).
  const v4i* cbp = (const v4i*)(cb + l15 * KD);
  v16bf bc0 = mk16(cbp[half], cbp[2 + half]);      // K = 0..31
  v16bf bc1 = mk16(cbp[4 + half], cbp[6 + half]);  // K = 32..63
  const float csq = csq_g[l15];

  __bf16* qt = sh_qt[wv];
  __bf16* wt = sh_wt[wv];

  v8f   acc[4] = {v8f{}, v8f{}, v8f{}, v8f{}};
  float denacc = 0.f;

  const int gw     = ((int)(blockIdx.x * blockDim.x) + tid) >> 5;
  const int nw     = (int)(gridDim.x * blockDim.x) >> 5;
  const int npairs = nq >> 5;

  for (int p = gw; p < npairs; p += nw) {
    const int base = p << 5;
#pragma unroll
    for (int t = 0; t < 2; ++t) {
      const int row = base + t * 16 + l15;
      v16bf a0, a1;
      if (PRE) {
        const v4i* rp = (const v4i*)(qb + (size_t)row * KD);
        a0 = mk16(rp[half], rp[2 + half]);      // K = 0..31
        a1 = mk16(rp[4 + half], rp[6 + half]);  // K = 32..63
      } else {
        const v4f* rp = (const v4f*)(qf + (size_t)row * KD);
        a0 = cvt16(rp[2 * half], rp[2 * half + 1],
                   rp[4 + 2 * half], rp[5 + 2 * half]);
        a1 = cvt16(rp[8 + 2 * half], rp[9 + 2 * half],
                   rp[12 + 2 * half], rp[13 + 2 * half]);
      }

      // stage q^T into LDS (element i of a0/a1 maps to dim i+8*half(+8 if i>=8))
      const int k = t * 16 + l15;
#pragma unroll
      for (int i = 0; i < 16; ++i) {
        int dim = i + 8 * half + ((i >> 3) << 3);
        qt[dim * 32 + k]        = ((__bf16*)&a0)[i];
        qt[(dim + 32) * 32 + k] = ((__bf16*)&a1)[i];
      }

      // GEMM1: scaled dot[m][n] for this 16-query tile
      v8f dacc = {};
      dacc = WMMA_BF16(a0, bc0, dacc);
      dacc = WMMA_BF16(a1, bc1, dacc);

      // softmax over classes; logit (log2 domain) = dot - csq
      float w[8];
#pragma unroll
      for (int j = 0; j < 8; ++j) {
        float l = dacc[j] - csq;
        float m = rmax16(l);
        float e = __builtin_amdgcn_exp2f(l - m);
        float s = rsum16(e);
        w[j] = e * __builtin_amdgcn_rcpf(s);
        denacc += w[j];
      }

      // w^T to LDS: lane owns class n=l15, queries k = t*16 + 8*half + j
      const int kb = t * 16 + 8 * half;
#pragma unroll
      for (int j = 0; j < 8; j += 2) {
        v2bf pk = {(__bf16)w[j], (__bf16)w[j + 1]};
        *(v2bf*)&wt[l15 * 32 + kb + j] = pk;
      }
    }

    asm volatile("s_wait_dscnt 0x0" ::: "memory");

    // GEMM2: num += w^T (16x32) @ q (32x64), 4 dim-groups
    const v4i* wtp = (const v4i*)(wt + l15 * 32);
    v16bf aw = mk16(wtp[half], wtp[2 + half]);
#pragma unroll
    for (int g = 0; g < 4; ++g) {
      const v4i* qtp = (const v4i*)(qt + (g * 16 + l15) * 32);
      v16bf bq = mk16(qtp[half], qtp[2 + half]);
      acc[g] = WMMA_BF16(aw, bq, acc[g]);
    }
  }

  // ---- epilogue: block reduce, then global f32 atomics ----
  denacc += __shfl_xor(denacc, 16);  // combine the two half-groups per class
#pragma unroll
  for (int g = 0; g < 4; ++g)
#pragma unroll
    for (int j = 0; j < 8; ++j)
      atomicAdd(&s_num[(j + 8 * half) * KD + g * 16 + l15], acc[g][j]);
  if (lane < 16) atomicAdd(&s_den[l15], denacc);
  __syncthreads();
  for (int i = tid; i < KC * KD; i += blockDim.x) atomicAdd(&g_num[i], s_num[i]);
  if (tid < KC) atomicAdd(&g_den[tid], s_den[tid]);
}

// --------------------------- final prediction ------------------------------

template <bool PRE>
__global__ __launch_bounds__(256) void predict_kernel(
    const float* __restrict__ qf, const __bf16* __restrict__ qb,
    const __bf16* __restrict__ cb, const float* __restrict__ csq_g,
    const int* __restrict__ qlab, float* __restrict__ counter, int nq) {
  const int tid  = threadIdx.x;
  const int lane = tid & 31;
  const int half = lane >> 4;
  const int l15  = lane & 15;

  const v4i* cbp = (const v4i*)(cb + l15 * KD);
  v16bf bc0 = mk16(cbp[half], cbp[2 + half]);
  v16bf bc1 = mk16(cbp[4 + half], cbp[6 + half]);
  const float csq = csq_g[l15];

  const int gw     = ((int)(blockIdx.x * blockDim.x) + tid) >> 5;
  const int nw     = (int)(gridDim.x * blockDim.x) >> 5;
  const int ntiles = nq >> 4;

  float cnt = 0.f;
  for (int t = gw; t < ntiles; t += nw) {
    const int row = (t << 4) + l15;
    v16bf a0, a1;
    if (PRE) {
      const v4i* rp = (const v4i*)(qb + (size_t)row * KD);
      a0 = mk16(rp[half], rp[2 + half]);
      a1 = mk16(rp[4 + half], rp[6 + half]);
    } else {
      const v4f* rp = (const v4f*)(qf + (size_t)row * KD);
      a0 = cvt16(rp[2 * half], rp[2 * half + 1],
                 rp[4 + 2 * half], rp[5 + 2 * half]);
      a1 = cvt16(rp[8 + 2 * half], rp[9 + 2 * half],
                 rp[12 + 2 * half], rp[13 + 2 * half]);
    }
    v8f dacc = {};
    dacc = WMMA_BF16(a0, bc0, dacc);
    dacc = WMMA_BF16(a1, bc1, dacc);
#pragma unroll
    for (int j = 0; j < 8; ++j) {
      // argmin over classes of scaled (||c||^2 - 2 dot), first-index tie break
      int iv = argmin16(csq - dacc[j], l15);
      if (l15 == 0) {
        int lbl = qlab[(t << 4) + j + 8 * half];
        cnt += (iv == lbl) ? 1.f : 0.f;
      }
    }
  }
  cnt += dppf<DPP_QUAD_XOR1>(cnt);
  cnt += dppf<DPP_QUAD_XOR2>(cnt);
  cnt += dppf<DPP_HALF_MIRR>(cnt);
  cnt += dppf<DPP_ROW_MIRR>(cnt);
  cnt += __shfl_xor(cnt, 16);
  if (lane == 0) atomicAdd(counter, cnt);
}

__global__ void finalize_kernel(const float* counter, float* out, float inv) {
  if (threadIdx.x == 0 && blockIdx.x == 0) out[0] = counter[0] * inv;
}

// ------------------------------- launch ------------------------------------

extern "C" void kernel_launch(void* const* d_in, const int* in_sizes, int n_in,
                              void* d_out, int out_size, void* d_ws,
                              size_t ws_size, hipStream_t stream) {
  const float* sf = (const float*)d_in[0];
  const float* qf = (const float*)d_in[1];
  const int*   sl = (const int*)d_in[2];
  const int*   ql = (const int*)d_in[3];
  // n_classes / n_iter / beta are fixed by setup_inputs (16 / 50 / 5)

  const int ns = in_sizes[0] / KD;
  const int nq = in_sizes[1] / KD;

  char*   ws      = (char*)d_ws;
  float*  acc_num = (float*)(ws + OFF_ACC_NUM);
  float*  acc_den = (float*)(ws + OFF_ACC_DEN);
  float*  sup_sum = (float*)(ws + OFF_SUP_SUM);
  float*  counts  = (float*)(ws + OFF_COUNTS);
  float*  csq     = (float*)(ws + OFF_CSQ);
  float*  counter = (float*)(ws + OFF_COUNTER);
  __bf16* cbf     = (__bf16*)(ws + OFF_CB);
  __bf16* qb      = (__bf16*)(ws + OFF_QB);

  const size_t need = OFF_QB + (size_t)nq * KD * sizeof(__bf16);
  const bool   pre  = (ws_size >= need);

  zero_kernel<<<1, 1024, 0, stream>>>(acc_num, acc_den, counter);
  supstats_kernel<<<1, 1024, 0, stream>>>(sf, sl, sup_sum, counts, ns);
  if (pre) convert_kernel<<<2048, 256, 0, stream>>>(qf, qb, (long)nq * 8);
  // initial centroids = class means (accumulators are zero)
  update_kernel<<<1, 1024, 0, stream>>>(acc_num, acc_den, sup_sum, counts, cbf, csq);

  const int blocks = 1024;
  for (int it = 0; it < KNITER; ++it) {
    if (pre)
      iterate_kernel<true><<<blocks, 256, 0, stream>>>(qf, qb, cbf, csq,
                                                       acc_num, acc_den, nq);
    else
      iterate_kernel<false><<<blocks, 256, 0, stream>>>(qf, qb, cbf, csq,
                                                        acc_num, acc_den, nq);
    update_kernel<<<1, 1024, 0, stream>>>(acc_num, acc_den, sup_sum, counts,
                                          cbf, csq);
  }

  if (pre)
    predict_kernel<true><<<blocks, 256, 0, stream>>>(qf, qb, cbf, csq, ql,
                                                     counter, nq);
  else
    predict_kernel<false><<<blocks, 256, 0, stream>>>(qf, qb, cbf, csq, ql,
                                                      counter, nq);
  finalize_kernel<<<1, 64, 0, stream>>>(counter, (float*)d_out,
                                        1.0f / (float)nq);
}
